// ChunkRetriever_91147795955933
// MI455X (gfx1250) — compile-verified, matrix operand order checked by reading, not run on hardware
//
#include <hip/hip_runtime.h>
#include <hip/hip_bf16.h>

typedef __attribute__((ext_vector_type(16))) __bf16 v16bf;
typedef __attribute__((ext_vector_type(8)))  __bf16 v8bf;
typedef __attribute__((ext_vector_type(8)))  float  v8f;

#define BATCH   4
#define L_DIM   4096
#define D_DIM   2048
#define RD      256
#define NCHUNK  64
#define TOPK    8
#define KVH     4
#define ROWS    32          // rows per workgroup
#define ASTRIDE 2056        // padded LDS stride for hidden tile (bf16 elems)
#define QSTRIDE 264         // padded LDS stride for q tile (bf16 elems)

// LDS carve-out (bytes)
#define OFF_A   0
#define OFF_Q   (OFF_A + ROWS * ASTRIDE * 2)
#define OFF_S   (OFF_Q + ROWS * QSTRIDE * 2)
#define OFF_RSQ (OFF_S + ROWS * NCHUNK * 4)
#define OFF_RST (OFF_RSQ + ROWS * 4)
#define OFF_QSQ (OFF_RST + ROWS * 4)
#define OFF_QRS (OFF_QSQ + ROWS * 4)
#define SMEM_BYTES (OFF_QRS + ROWS * 4)

// ---------------------------------------------------------------------------
// Load a 16x32 bf16 WMMA fragment (A layout; B 32x16 mirrors it with lane=col).
// lane i (0..15): row = row0+i, K elems [k0..k0+7] and [k0+16..k0+23]
// lane i+16     : row = row0+i, K elems [k0+8..k0+15] and [k0+24..k0+31]
// ---------------------------------------------------------------------------
__device__ __forceinline__ v16bf frag_ld(const __bf16* base, int row0, int stride,
                                         int k0, int half, int li) {
    const __bf16* p = base + (long)(row0 + li) * stride + k0 + half * 8;
    union { v16bf v; v8bf h[2]; } u;
    u.h[0] = *(const v8bf*)(p);
    u.h[1] = *(const v8bf*)(p + 16);
    return u.v;
}

__device__ __forceinline__ v8f wmma_bf16(v16bf a, v16bf b, v8f c) {
    return __builtin_amdgcn_wmma_f32_16x16x32_bf16(false, a, false, b,
                                                   (short)0, c, false, false);
}

// ---------------------------------------------------------------------------
// Prep 1: Wn[r,d] = q_proj_w[r,d] * pre_norm_w[d]   (bf16)
// ---------------------------------------------------------------------------
__global__ void prep_weight(const float* __restrict__ W,
                            const float* __restrict__ prew,
                            __bf16* __restrict__ Wn) {
    int i = blockIdx.x * blockDim.x + threadIdx.x;   // 0 .. RD*D_DIM-1
    Wn[i] = (__bf16)(W[i] * prew[i & (D_DIM - 1)]);
}

// ---------------------------------------------------------------------------
// Prep 2: lmkw[b,c,r] = rmsnorm(lmk[b,c,:])[r] * lmk_norm_w[r] * q_norm_w[r]
// One 256-thread block per (b,c) row.
// ---------------------------------------------------------------------------
__global__ void prep_lmk(const float* __restrict__ lmk,
                         const float* __restrict__ lmk_w,
                         const float* __restrict__ q_w,
                         __bf16* __restrict__ out) {
    __shared__ float red[RD];
    int row = blockIdx.x;         // b*NCHUNK + c
    int r   = threadIdx.x;        // 0..255
    float v = lmk[(size_t)row * RD + r];
    red[r] = v * v;
    __syncthreads();
    for (int s = RD / 2; s > 0; s >>= 1) {
        if (r < s) red[r] += red[r + s];
        __syncthreads();
    }
    float rs = rsqrtf(red[0] / (float)RD + 1e-5f);
    out[(size_t)row * RD + r] = (__bf16)(v * rs * lmk_w[r] * q_w[r]);
}

// ---------------------------------------------------------------------------
// Fused main kernel: 1 block = 32 rows of one batch.
//   stage 1: hidden -> bf16 LDS tile + row sum(x^2)
//   stage 2: GEMM1 (WMMA bf16) -> q, scaled by rsqrt, bf16 -> LDS
//   stage 3: row sum(q^2) -> per-row score scale (folds q-rms and 1/sqrt(256))
//   stage 4: GEMM2 (WMMA bf16) -> 32x64 scores in LDS
//   stage 5: causal mask + top-8 + sort + softmax + broadcast writes
// ---------------------------------------------------------------------------
__global__ void chunk_retriever_main(const float* __restrict__ hidden,
                                     const __bf16* __restrict__ Wn,
                                     const __bf16* __restrict__ lmkw,
                                     float* __restrict__ w_out,
                                     int*   __restrict__ i_out) {
    extern __shared__ char smem[];
    __bf16* aTile = (__bf16*)(smem + OFF_A);
    __bf16* qTile = (__bf16*)(smem + OFF_Q);
    float*  sTile = (float*)(smem + OFF_S);
    float*  rowsq = (float*)(smem + OFF_RSQ);
    float*  rstd  = (float*)(smem + OFF_RST);
    float*  qsq   = (float*)(smem + OFF_QSQ);
    float*  qrs   = (float*)(smem + OFF_QRS);

    const int tid  = threadIdx.x;           // 0..255
    const int g0   = blockIdx.x * ROWS;     // flattened (b*L + l) row base
    const int bidx = g0 / L_DIM;
    const int pos0 = g0 % L_DIM;

    // ---------------- stage 1: load hidden, sum squares, bf16 to LDS -------
    if (tid < ROWS) { rowsq[tid] = 0.f; qsq[tid] = 0.f; }
    __syncthreads();
    {
        const int r    = tid >> 3;          // 0..31 (8 threads per row)
        const int cseg = (tid & 7) * 256;   // 256-col segment
        const float* hrow = hidden + (size_t)(g0 + r) * D_DIM + cseg;
        __bf16* dst = aTile + r * ASTRIDE + cseg;
        float ss = 0.f;
        for (int i = 0; i < 256; i += 4) {
            float4 v = *(const float4*)(hrow + i);
            ss += v.x * v.x + v.y * v.y + v.z * v.z + v.w * v.w;
            dst[i + 0] = (__bf16)v.x;
            dst[i + 1] = (__bf16)v.y;
            dst[i + 2] = (__bf16)v.z;
            dst[i + 3] = (__bf16)v.w;
        }
        atomicAdd(&rowsq[r], ss);
    }
    __syncthreads();
    if (tid < ROWS) rstd[tid] = rsqrtf(rowsq[tid] / (float)D_DIM + 1e-5f);
    __syncthreads();

    // ---------------- stage 2: GEMM1  q = x * Wn^T --------------------------
    const int wave = tid >> 5;
    const int lane = tid & 31;
    const int half = lane >> 4;
    const int li   = lane & 15;
    {
        v8f acc[2][2];
        acc[0][0] = (v8f){}; acc[0][1] = (v8f){};
        acc[1][0] = (v8f){}; acc[1][1] = (v8f){};
        const int n0 = wave * 32;           // this wave's 32 output columns
        for (int k0 = 0; k0 < D_DIM; k0 += 32) {
            v16bf a0 = frag_ld(aTile, 0,        ASTRIDE, k0, half, li);
            v16bf a1 = frag_ld(aTile, 16,       ASTRIDE, k0, half, li);
            v16bf b0 = frag_ld(Wn,    n0,       D_DIM,   k0, half, li);
            v16bf b1 = frag_ld(Wn,    n0 + 16,  D_DIM,   k0, half, li);
            acc[0][0] = wmma_bf16(a0, b0, acc[0][0]);
            acc[0][1] = wmma_bf16(a0, b1, acc[0][1]);
            acc[1][0] = wmma_bf16(a1, b0, acc[1][0]);
            acc[1][1] = wmma_bf16(a1, b1, acc[1][1]);
        }
        // scale by pre-norm rsqrt, store bf16 q to LDS
        for (int m = 0; m < 2; ++m)
            for (int n = 0; n < 2; ++n)
                for (int j = 0; j < 8; ++j) {
                    int rr = m * 16 + j + half * 8;   // row within block tile
                    float v = acc[m][n][j] * rstd[rr];
                    qTile[rr * QSTRIDE + n0 + n * 16 + li] = (__bf16)v;
                }
    }
    __syncthreads();

    // ---------------- stage 3: q row rms (fold 1/sqrt(RD) into scale) ------
    {
        const int rr = tid >> 3;
        const int c0 = (tid & 7) * 32;
        float ss = 0.f;
        for (int i = 0; i < 32; ++i) {
            float v = (float)qTile[rr * QSTRIDE + c0 + i];
            ss += v * v;
        }
        atomicAdd(&qsq[rr], ss);
    }
    __syncthreads();
    if (tid < ROWS)
        qrs[tid] = rsqrtf(qsq[tid] / (float)RD + 1e-5f) * (1.0f / 16.0f);
    __syncthreads();

    // ---------------- stage 4: GEMM2  scores = q * lmkw^T -------------------
    {
        const int m = wave >> 2;            // 0..1 -> row tile
        const int n = wave & 3;             // 0..3 -> chunk tile
        const __bf16* lmkB = lmkw + (size_t)bidx * NCHUNK * RD;
        v8f acc = (v8f){};
        for (int k0 = 0; k0 < RD; k0 += 32) {
            v16bf a = frag_ld(qTile, m * 16, QSTRIDE, k0, half, li);
            v16bf b = frag_ld(lmkB,  n * 16, RD,      k0, half, li);
            acc = wmma_bf16(a, b, acc);
        }
        for (int j = 0; j < 8; ++j) {
            int rr = m * 16 + j + half * 8;
            sTile[rr * NCHUNK + n * 16 + li] = acc[j] * qrs[rr];
        }
    }
    __syncthreads();

    // ---------------- stage 5: mask + top-8 + softmax + writes --------------
    if (tid < ROWS) {
        const int row = tid;
        const int pos = pos0 + row;
        int nv = pos >> 6;                  // number of causally valid chunks
        if (nv > NCHUNK) nv = NCHUNK;
        const float NEG = -__builtin_inff();
        const float* srow = sTile + row * NCHUNK;

        unsigned long long picked = 0ull;
        int   idxs[TOPK];
        float vals[TOPK];
        for (int t = 0; t < TOPK; ++t) {
            float best = NEG;
            int   bi   = -1;
            for (int c = 0; c < NCHUNK; ++c) {
                if ((picked >> c) & 1ull) continue;
                float v = (c < nv) ? srow[c] : NEG;
                if (v > best || bi < 0) { best = v; bi = c; }
            }
            picked |= (1ull << bi);
            idxs[t] = bi;
            vals[t] = best;
        }
        // sort pairs by index ascending (insertion sort, 8 elems)
        for (int i = 1; i < TOPK; ++i) {
            int   ki = idxs[i];
            float kv = vals[i];
            int j = i - 1;
            while (j >= 0 && idxs[j] > ki) {
                idxs[j + 1] = idxs[j];
                vals[j + 1] = vals[j];
                --j;
            }
            idxs[j + 1] = ki;
            vals[j + 1] = kv;
        }
        // softmax (-inf entries -> 0; all -inf -> all zeros)
        float wts[TOPK];
        float mx = NEG;
        for (int t = 0; t < TOPK; ++t) mx = fmaxf(mx, vals[t]);
        if (mx == NEG) {
            for (int t = 0; t < TOPK; ++t) wts[t] = 0.f;
        } else {
            float s = 0.f;
            for (int t = 0; t < TOPK; ++t) {
                float e = (vals[t] == NEG) ? 0.f : __expf(vals[t] - mx);
                wts[t] = e;
                s += e;
            }
            float inv = 1.0f / s;
            for (int t = 0; t < TOPK; ++t) wts[t] *= inv;
        }
        // broadcast over KV heads
        size_t obase = ((size_t)(bidx * L_DIM + pos) * KVH) * TOPK;
        for (int h = 0; h < KVH; ++h)
            for (int t = 0; t < TOPK; ++t) {
                w_out[obase + h * TOPK + t] = wts[t];
                i_out[obase + h * TOPK + t] = idxs[t];
            }
    }
}

// ---------------------------------------------------------------------------
extern "C" void kernel_launch(void* const* d_in, const int* in_sizes, int n_in,
                              void* d_out, int out_size, void* d_ws, size_t ws_size,
                              hipStream_t stream) {
    const float* hidden     = (const float*)d_in[0];
    const float* landmarks  = (const float*)d_in[1];
    const float* q_proj_w   = (const float*)d_in[2];
    const float* pre_norm_w = (const float*)d_in[3];
    const float* q_norm_w   = (const float*)d_in[4];
    const float* lmk_norm_w = (const float*)d_in[5];

    __bf16* Wn   = (__bf16*)d_ws;                                  // 1 MB
    __bf16* lmkw = (__bf16*)((char*)d_ws + (size_t)RD * D_DIM * 2); // 128 KB

    float* w_out = (float*)d_out;
    int*   i_out = (int*)(w_out + (size_t)BATCH * L_DIM * KVH * TOPK);

    // Prep: fold pre_norm into weights; normalize landmarks w/ folded scales.
    prep_weight<<<(RD * D_DIM) / 256, 256, 0, stream>>>(q_proj_w, pre_norm_w, Wn);
    prep_lmk<<<BATCH * NCHUNK, RD, 0, stream>>>(landmarks, lmk_norm_w, q_norm_w, lmkw);

    // Fused GEMM1 + GEMM2 + top-k/softmax: one block per 32 rows.
    chunk_retriever_main<<<(BATCH * L_DIM) / ROWS, 256, SMEM_BYTES, stream>>>(
        hidden, Wn, lmkw, w_out, i_out);
}
